// Encoder_10239202034097
// MI455X (gfx1250) — matrix operand order repaired and checked by reading, not compiled, not run on hardware
//
#include <hip/hip_runtime.h>

// ---------------------------------------------------------------------------
// Problem constants (from reference)
// ---------------------------------------------------------------------------
#define N_NODES  50000
#define N_EDGES  400000
#define DIM      512
#define N_LAYERS 4
#define N_GRAPHS 500
#define EPS_F    1e-5f
#define NEG_SLOPE 0.01f

#define MT 5              // 5 x 16 = 80 rows per block (50000 = 625 * 80)
#define NT 4              // 4 x 16 = 64 cols per wave  (8 waves -> 512 cols)
#define ROWS_PER_BLOCK (MT * 16)

// ---------------------------------------------------------------------------
// Types
// ---------------------------------------------------------------------------
typedef __attribute__((ext_vector_type(16))) __bf16        v16bf;
typedef __attribute__((ext_vector_type(8)))  float         v8f;
typedef __attribute__((ext_vector_type(4)))  float         f32x4;
typedef __attribute__((ext_vector_type(4)))  unsigned int  u32x4;
typedef __attribute__((ext_vector_type(2)))  unsigned int  u32x2;

union FragBF {          // 32 bytes: one WMMA bf16 A/B fragment per lane
    v16bf v;
    u32x4 q[2];
};

__device__ __forceinline__ unsigned short f32_to_bf16(float f) {
    union { float f; unsigned u; } v; v.f = f;
    unsigned r = v.u + 0x7FFFu + ((v.u >> 16) & 1u);   // round-to-nearest-even
    return (unsigned short)(r >> 16);
}

__device__ __forceinline__ float leaky(float v) {
    return v >= 0.0f ? v : NEG_SLOPE * v;
}

// ---------------------------------------------------------------------------
// Utility kernels
// ---------------------------------------------------------------------------
__global__ void zero_f32(float* p, int n) {
    int i = blockIdx.x * blockDim.x + threadIdx.x;
    if (i < n) p[i] = 0.0f;
}

// z[n][0..511] = h[n*hstride + 0..511]   (strided row gather, vectorized x4)
__global__ void copy_rows(const float* __restrict__ h, int hstride, float* __restrict__ z) {
    int i = blockIdx.x * blockDim.x + threadIdx.x;       // over N*128 float4s
    if (i >= N_NODES * (DIM / 4)) return;
    int n  = i >> 7;
    int c4 = (i & 127) << 2;
    const f32x4 v = *(const f32x4*)(h + (size_t)n * hstride + c4);
    *(f32x4*)(z + (size_t)n * DIM + c4) = v;
}

// z[dst] += h[src] for every edge (4 floats per thread)
__global__ void scatter_add(const float* __restrict__ h, int hstride,
                            const int* __restrict__ ei, float* __restrict__ z) {
    long long gid = (long long)blockIdx.x * blockDim.x + threadIdx.x;
    if (gid >= (long long)N_EDGES * (DIM / 4)) return;
    int e  = (int)(gid >> 7);
    int c4 = (int)(gid & 127) << 2;
    int src = ei[e];
    int dst = ei[N_EDGES + e];
    const f32x4 v = *(const f32x4*)(h + (size_t)src * hstride + c4);
    float* o = z + (size_t)dst * DIM + c4;
    atomicAdd(o + 0, v.x);
    atomicAdd(o + 1, v.y);
    atomicAdd(o + 2, v.z);
    atomicAdd(o + 3, v.w);
}

// zb = bf16(z), 4 elements per thread
__global__ void cvt_bf16(const float* __restrict__ z, unsigned short* __restrict__ zb) {
    int i = blockIdx.x * blockDim.x + threadIdx.x;       // over N*128
    if (i >= N_NODES * (DIM / 4)) return;
    const f32x4 v = ((const f32x4*)z)[i];
    u32x2 o;
    o.x = (unsigned)f32_to_bf16(v.x) | ((unsigned)f32_to_bf16(v.y) << 16);
    o.y = (unsigned)f32_to_bf16(v.z) | ((unsigned)f32_to_bf16(v.w) << 16);
    ((u32x2*)zb)[i] = o;
}

// Pack a 512x512 row-major f32 weight matrix [k][n] into per-lane WMMA
// B-fragment order: Wp[ ((kt*512 + n)*2 + half)*16 + j ] = bf16(W[kt*32 + half*16 + j][n])
// -> each lane of a wave reads a contiguous 32B chunk per k-step.
__global__ void pack_w(const float* __restrict__ W, unsigned short* __restrict__ Wp) {
    int idx = blockIdx.x * blockDim.x + threadIdx.x;     // over 512*512
    if (idx >= DIM * DIM) return;
    int j    = idx & 15;
    int half = (idx >> 4) & 1;
    int n    = (idx >> 5) & 511;
    int kt   = idx >> 14;
    int k    = kt * 32 + half * 16 + j;
    Wp[idx] = f32_to_bf16(W[(size_t)k * DIM + n]);
}

// ---------------------------------------------------------------------------
// WMMA GEMM core: one block = 80 rows x 512 cols of the output.
// 8 waves; wave w owns all 80 rows x cols [w*64, w*64+64):
//   5 (M) x 4 (N) f32 accumulators, K-loop of 16 steps of K=32,
//   register double-buffered A/B fragments (prefetch kt+1 before kt's WMMAs).
//
// A fragment (16x32 bf16, row-major src): lane l holds row m = base + (l&15),
//   K = (l>>4)*16 + j, j = 0..15  -> two adjacent b128 loads.
// B fragment: lane l holds col n = colBase + (l&15), K = (l>>4)*16 + j,
//   pre-packed contiguous 32B.
// ---------------------------------------------------------------------------
__device__ __forceinline__ void load_a(FragBF a[MT], const u32x4* __restrict__ Abase,
                                       int kt, int half) {
    const int q0 = kt * 4 + half * 2;    // (kt*32 + half*16) / 8
#pragma unroll
    for (int mt = 0; mt < MT; ++mt) {
        const u32x4* row = Abase + (size_t)mt * 16 * (DIM / 8);
        a[mt].q[0] = row[q0];
        a[mt].q[1] = row[q0 + 1];
    }
}

__device__ __forceinline__ void load_b(FragBF b[NT], const unsigned short* __restrict__ Bp,
                                       int kt, int colBase, int lane) {
    const int half = lane >> 4;
#pragma unroll
    for (int nt = 0; nt < NT; ++nt) {
        int n = colBase + nt * 16 + (lane & 15);
        const u32x4* bp = (const u32x4*)(Bp + ((((size_t)kt * DIM + n) * 2 + half) << 4));
        b[nt].q[0] = bp[0];
        b[nt].q[1] = bp[1];
    }
}

__device__ __forceinline__ void mma_step(const FragBF a[MT], const FragBF b[NT],
                                         v8f acc[MT][NT]) {
#pragma unroll
    for (int mt = 0; mt < MT; ++mt)
#pragma unroll
        for (int nt = 0; nt < NT; ++nt)
            acc[mt][nt] = __builtin_amdgcn_wmma_f32_16x16x32_bf16(
                false, a[mt].v, false, b[nt].v, (short)0, acc[mt][nt], false, false);
}

__device__ __forceinline__ void wmma_tile(const unsigned short* __restrict__ A,
                                          const unsigned short* __restrict__ Bp,
                                          int rowBase, int colBase, int lane,
                                          v8f acc[MT][NT]) {
    const int half = lane >> 4;
    const u32x4* Abase = (const u32x4*)(A + (size_t)(rowBase + (lane & 15)) * DIM);

    FragBF a0[MT], b0[NT], a1[MT], b1[NT];
    load_a(a0, Abase, 0, half);
    load_b(b0, Bp, 0, colBase, lane);

    for (int kt = 0; kt < 16; kt += 2) {
        // prefetch kt+1 while kt's WMMAs run
        load_a(a1, Abase, kt + 1, half);
        load_b(b1, Bp, kt + 1, colBase, lane);
        mma_step(a0, b0, acc);
        // prefetch kt+2 while kt+1's WMMAs run
        if (kt + 2 < 16) {
            load_a(a0, Abase, kt + 2, half);
            load_b(b0, Bp, kt + 2, colBase, lane);
        }
        mma_step(a1, b1, acc);
    }
}

// GEMM1: y = bf16( leaky(zb @ W1 + b1) )
__global__ __launch_bounds__(256) void gemm1_kernel(const unsigned short* __restrict__ zb,
                                                    const unsigned short* __restrict__ Wp,
                                                    const float* __restrict__ bias,
                                                    unsigned short* __restrict__ y) {
    const int lane = threadIdx.x & 31;
    const int wave = threadIdx.x >> 5;
    const int rowBase = blockIdx.x * ROWS_PER_BLOCK;
    const int colBase = wave * (NT * 16);
    const int half = lane >> 4;

    v8f acc[MT][NT] = {};
    wmma_tile(zb, Wp, rowBase, colBase, lane, acc);

    float bv[NT];
#pragma unroll
    for (int nt = 0; nt < NT; ++nt)
        bv[nt] = bias[colBase + nt * 16 + (lane & 15)];

#pragma unroll
    for (int mt = 0; mt < MT; ++mt)
#pragma unroll
        for (int nt = 0; nt < NT; ++nt) {
            int col = colBase + nt * 16 + (lane & 15);
#pragma unroll
            for (int r = 0; r < 8; ++r) {
                float v = leaky(acc[mt][nt][r] + bv[nt]);
                int row = rowBase + mt * 16 + r + half * 8;
                y[(size_t)row * DIM + col] = f32_to_bf16(v);
            }
        }
}

// GEMM2 + fused epilogue:
//   t = NodeNorm_row( leaky(y @ W2 + b2) )   (row mean/var via LDS reduction)
//   also accumulates per-column sum / sumsq (global atomics) for BatchNorm.
__global__ __launch_bounds__(256) void gemm2_kernel(const unsigned short* __restrict__ yb,
                                                    const unsigned short* __restrict__ Wp,
                                                    const float* __restrict__ bias,
                                                    float* __restrict__ tout,
                                                    float* __restrict__ colsum,
                                                    float* __restrict__ colsq) {
    __shared__ float rsum[ROWS_PER_BLOCK];
    __shared__ float rsq[ROWS_PER_BLOCK];
    const int lane = threadIdx.x & 31;
    const int wave = threadIdx.x >> 5;
    const int rowBase = blockIdx.x * ROWS_PER_BLOCK;
    const int colBase = wave * (NT * 16);
    const int half = lane >> 4;

    if (threadIdx.x < ROWS_PER_BLOCK) { rsum[threadIdx.x] = 0.0f; rsq[threadIdx.x] = 0.0f; }
    __syncthreads();

    v8f acc[MT][NT] = {};
    wmma_tile(yb, Wp, rowBase, colBase, lane, acc);

    float bv[NT];
#pragma unroll
    for (int nt = 0; nt < NT; ++nt)
        bv[nt] = bias[colBase + nt * 16 + (lane & 15)];

    // bias + leaky in place, then per-row sum/sumsq partials (lane covers 4 cols)
#pragma unroll
    for (int mt = 0; mt < MT; ++mt)
#pragma unroll
        for (int r = 0; r < 8; ++r) {
            float s = 0.0f, q = 0.0f;
#pragma unroll
            for (int nt = 0; nt < NT; ++nt) {
                float v = leaky(acc[mt][nt][r] + bv[nt]);
                acc[mt][nt][r] = v;
                s += v; q += v * v;
            }
            int row = mt * 16 + r + half * 8;
            atomicAdd(&rsum[row], s);
            atomicAdd(&rsq[row], q);
        }
    __syncthreads();

    // normalize per row, store t, accumulate per-column stats (40 rows per lane-col)
    float cs[NT] = {}, cq[NT] = {};
#pragma unroll
    for (int mt = 0; mt < MT; ++mt) {
        float mean8[8], inv8[8];
#pragma unroll
        for (int r = 0; r < 8; ++r) {
            int row = mt * 16 + r + half * 8;
            float mean = rsum[row] * (1.0f / DIM);
            float var  = rsq[row] * (1.0f / DIM) - mean * mean;
            mean8[r] = mean;
            inv8[r]  = rsqrtf(var + EPS_F);
        }
#pragma unroll
        for (int nt = 0; nt < NT; ++nt) {
            int col = colBase + nt * 16 + (lane & 15);
#pragma unroll
            for (int r = 0; r < 8; ++r) {
                float v = (acc[mt][nt][r] - mean8[r]) * inv8[r];
                cs[nt] += v; cq[nt] += v * v;
                int row = rowBase + mt * 16 + r + half * 8;
                tout[(size_t)row * DIM + col] = v;
            }
        }
    }
#pragma unroll
    for (int nt = 0; nt < NT; ++nt) {
        int col = colBase + nt * 16 + (lane & 15);
        atomicAdd(&colsum[col], cs[nt]);
        atomicAdd(&colsq[col], cq[nt]);
    }
}

// BatchNorm apply + write x_local slice (xl points at x_local + layer*512,
// row stride 2048). Vectorized x4.
__global__ void bn_apply(const float* __restrict__ t,
                         const float* __restrict__ colsum,
                         const float* __restrict__ colsq,
                         const float* __restrict__ gamma,
                         const float* __restrict__ beta,
                         float* __restrict__ xl) {
    int i = blockIdx.x * blockDim.x + threadIdx.x;       // over N*128
    if (i >= N_NODES * (DIM / 4)) return;
    int row = i >> 7;
    int c4  = (i & 127) << 2;
    f32x4 v = ((const f32x4*)t)[i];
    f32x4 o;
#pragma unroll
    for (int k = 0; k < 4; ++k) {
        int c = c4 + k;
        float mean = colsum[c] * (1.0f / N_NODES);
        float var  = colsq[c] * (1.0f / N_NODES) - mean * mean;
        o[k] = (v[k] - mean) * rsqrtf(var + EPS_F) * gamma[c] + beta[c];
    }
    *(f32x4*)(xl + (size_t)row * (N_LAYERS * DIM) + c4) = o;
}

// global_add_pool over all 4 concatenated layer slices at once
__global__ void pool_kernel(const float* __restrict__ xl,
                            const int* __restrict__ batch,
                            float* __restrict__ xg) {
    long long gid = (long long)blockIdx.x * blockDim.x + threadIdx.x;  // N * 512 (x4 cols)
    if (gid >= (long long)N_NODES * (N_LAYERS * DIM / 4)) return;
    int n  = (int)(gid >> 9);
    int c4 = ((int)gid & 511) << 2;
    int g  = batch[n];
    const f32x4 v = *(const f32x4*)(xl + (size_t)n * (N_LAYERS * DIM) + c4);
    float* o = xg + (size_t)g * (N_LAYERS * DIM) + c4;
    atomicAdd(o + 0, v.x);
    atomicAdd(o + 1, v.y);
    atomicAdd(o + 2, v.z);
    atomicAdd(o + 3, v.w);
}

// ---------------------------------------------------------------------------
// Host launch
// ---------------------------------------------------------------------------
static inline int cdiv(long long a, int b) { return (int)((a + b - 1) / b); }

extern "C" void kernel_launch(void* const* d_in, const int* in_sizes, int n_in,
                              void* d_out, int out_size, void* d_ws, size_t ws_size,
                              hipStream_t stream) {
    const float* x     = (const float*)d_in[0];
    const int*   ei    = (const int*)d_in[1];     // [2, E]
    const int*   batch = (const int*)d_in[2];     // [N]
    const float* W1    = (const float*)d_in[3];   // [L, D, D]
    const float* b1    = (const float*)d_in[4];   // [L, D]
    const float* W2    = (const float*)d_in[5];
    const float* b2    = (const float*)d_in[6];
    const float* gamma = (const float*)d_in[7];
    const float* beta  = (const float*)d_in[8];

    float* xg = (float*)d_out;                                     // [G, L*D]
    float* xl = (float*)d_out + (size_t)N_GRAPHS * N_LAYERS * DIM; // [N, L*D]

    // workspace carve-up
    char* ws = (char*)d_ws;
    float* z = (float*)ws;                       ws += (size_t)N_NODES * DIM * 4;  // aggr / t (reused)
    unsigned short* zb = (unsigned short*)ws;    ws += (size_t)N_NODES * DIM * 2;  // bf16 activations
    unsigned short* yb = (unsigned short*)ws;    ws += (size_t)N_NODES * DIM * 2;  // bf16 mid-MLP
    unsigned short* Wp = (unsigned short*)ws;    ws += (size_t)2 * N_LAYERS * DIM * DIM * 2; // packed W
    float* colstats = (float*)ws;                // 1024 floats (sum | sumsq)

    const size_t WSZ = (size_t)DIM * DIM;        // 262144 elements per matrix

    // pack all weights into WMMA B-fragment order (cheap, done per launch)
    for (int l = 0; l < N_LAYERS; ++l) {
        pack_w<<<cdiv(WSZ, 256), 256, 0, stream>>>(W1 + (size_t)l * WSZ, Wp + (size_t)(2 * l) * WSZ);
        pack_w<<<cdiv(WSZ, 256), 256, 0, stream>>>(W2 + (size_t)l * WSZ, Wp + (size_t)(2 * l + 1) * WSZ);
    }

    // zero the pooled-output region of d_out
    zero_f32<<<cdiv((long long)N_GRAPHS * N_LAYERS * DIM, 256), 256, 0, stream>>>(
        xg, N_GRAPHS * N_LAYERS * DIM);

    const int elem4 = N_NODES * (DIM / 4);
    const int edge4 = (int)((long long)N_EDGES * (DIM / 4));
    const int gemm_grid = N_NODES / ROWS_PER_BLOCK;    // 625

    for (int l = 0; l < N_LAYERS; ++l) {
        const float* h  = (l == 0) ? x : (xl + (size_t)(l - 1) * DIM);
        const int hstr  = (l == 0) ? DIM : (N_LAYERS * DIM);

        zero_f32<<<4, 256, 0, stream>>>(colstats, 1024);

        // z = h ; z[dst] += h[src]
        copy_rows<<<cdiv(elem4, 256), 256, 0, stream>>>(h, hstr, z);
        scatter_add<<<cdiv(edge4, 256), 256, 0, stream>>>(h, hstr, ei, z);

        // bf16 convert for WMMA A-operand
        cvt_bf16<<<cdiv(elem4, 256), 256, 0, stream>>>(z, zb);

        // MLP: y = leaky(z@W1+b1) ; t = NodeNorm(leaky(y@W2+b2)) + BN stats
        gemm1_kernel<<<gemm_grid, 256, 0, stream>>>(
            zb, Wp + (size_t)(2 * l) * WSZ, b1 + (size_t)l * DIM, yb);
        gemm2_kernel<<<gemm_grid, 256, 0, stream>>>(
            yb, Wp + (size_t)(2 * l + 1) * WSZ, b2 + (size_t)l * DIM,
            z /* t reuses z */, colstats, colstats + DIM);

        // BatchNorm apply -> x_local slice (doubles as next-layer h)
        bn_apply<<<cdiv(elem4, 256), 256, 0, stream>>>(
            z, colstats, colstats + DIM, gamma + (size_t)l * DIM, beta + (size_t)l * DIM,
            xl + (size_t)l * DIM);
    }

    // global_add_pool over concatenated features
    pool_kernel<<<cdiv((long long)N_NODES * (N_LAYERS * DIM / 4), 256), 256, 0, stream>>>(
        xl, batch, xg);
}
